// M_1803886265139
// MI455X (gfx1250) — matrix operand order repaired and checked by reading, not compiled, not run on hardware
//
#include <hip/hip_runtime.h>
#include <hip/hip_bf16.h>
#include <math.h>

typedef _Float16 v16h __attribute__((ext_vector_type(16)));
typedef _Float16 v8h  __attribute__((ext_vector_type(8)));
typedef float    v8f  __attribute__((ext_vector_type(8)));

#define D_DIM   256          // K
#define F_DIM   1024         // N (output features)
#define M_TILE  64           // rows per block
#define N_CHUNK 128          // W columns staged in LDS per iteration
#define LDSW    264          // padded row stride in halfs (132 dwords -> 4 mod 64 banks)

__global__ __launch_bounds__(256, 2)
void fused_ln_gemm_gelu(const float* __restrict__ x399,
                        const float* __restrict__ x365,
                        const float* __restrict__ lng,
                        const float* __restrict__ lnb,
                        const float* __restrict__ W,
                        const float* __restrict__ bias,
                        float* __restrict__ out)
{
    __shared__ __attribute__((aligned(16))) _Float16 Alds[M_TILE * LDSW];   // ~33.8 KB
    __shared__ __attribute__((aligned(16))) _Float16 Blds[N_CHUNK * LDSW];  // ~67.6 KB

    const int tid   = threadIdx.x;
    const int lane  = tid & 31;
    const int wave  = tid >> 5;
    const int half  = lane >> 4;   // 0 or 1
    const int l16   = lane & 15;
    const int rowBase = blockIdx.x * M_TILE;

    // ---------- Phase 1: residual add + LayerNorm -> f16 tile in Alds ----------
    // wave handles 8 rows; lane covers 8 contiguous columns (c0..c0+7)
    {
        const int c0 = lane * 8;
        float4 g0 = *(const float4*)(lng + c0);
        float4 g1 = *(const float4*)(lng + c0 + 4);
        float4 e0 = *(const float4*)(lnb + c0);
        float4 e1 = *(const float4*)(lnb + c0 + 4);
        float gs[8] = {g0.x,g0.y,g0.z,g0.w,g1.x,g1.y,g1.z,g1.w};
        float es[8] = {e0.x,e0.y,e0.z,e0.w,e1.x,e1.y,e1.z,e1.w};

        #pragma unroll
        for (int r = 0; r < 8; ++r) {
            const int row = wave * 8 + r;
            const long goff = (long)(rowBase + row) * D_DIM + c0;
            float4 a0 = *(const float4*)(x399 + goff);
            float4 a1 = *(const float4*)(x399 + goff + 4);
            float4 b0 = *(const float4*)(x365 + goff);
            float4 b1 = *(const float4*)(x365 + goff + 4);
            float v[8];
            v[0]=a0.x+b0.x; v[1]=a0.y+b0.y; v[2]=a0.z+b0.z; v[3]=a0.w+b0.w;
            v[4]=a1.x+b1.x; v[5]=a1.y+b1.y; v[6]=a1.z+b1.z; v[7]=a1.w+b1.w;

            float sum = 0.f, sq = 0.f;
            #pragma unroll
            for (int j = 0; j < 8; ++j) { sum += v[j]; sq += v[j]*v[j]; }
            #pragma unroll
            for (int m = 16; m >= 1; m >>= 1) {
                sum += __shfl_xor(sum, m, 32);
                sq  += __shfl_xor(sq,  m, 32);
            }
            const float mu   = sum * (1.0f / 256.0f);
            const float var  = sq  * (1.0f / 256.0f) - mu * mu;
            const float rstd = rsqrtf(var + 1e-12f);

            v8h h;
            #pragma unroll
            for (int j = 0; j < 8; ++j)
                h[j] = (_Float16)((v[j] - mu) * rstd * gs[j] + es[j]);
            *(v8h*)(&Alds[row * LDSW + c0]) = h;
        }
    }

    const int waveM = wave & 3;   // M sub-tile 0..3
    const int waveN = wave >> 2;  // N half within chunk: 0..1

    // ---------- Phase 2: loop over the 8 N-chunks of W ----------
    for (int nc = 0; nc < F_DIM / N_CHUNK; ++nc) {
        __syncthreads();  // A ready / previous chunk's compute done

        // stage W chunk (fp32 -> f16) into Blds: thread pair per row
        {
            const int n  = tid >> 1;               // 0..127
            const int kh = (tid & 1) * 128;        // half-row
            const long wbase = (long)(nc * N_CHUNK + n) * D_DIM + kh;
            #pragma unroll
            for (int j = 0; j < 16; ++j) {
                float4 wa = *(const float4*)(W + wbase + j * 8);
                float4 wb = *(const float4*)(W + wbase + j * 8 + 4);
                v8h h;
                h[0]=(_Float16)wa.x; h[1]=(_Float16)wa.y; h[2]=(_Float16)wa.z; h[3]=(_Float16)wa.w;
                h[4]=(_Float16)wb.x; h[5]=(_Float16)wb.y; h[6]=(_Float16)wb.z; h[7]=(_Float16)wb.w;
                *(v8h*)(&Blds[n * LDSW + kh + j * 8]) = h;
            }
        }
        __syncthreads();

        v8f acc[4] = {};   // 4 x (16x16 f32) accumulators

        #pragma unroll
        for (int kk = 0; kk < D_DIM; kk += 32) {
            // A fragment: lane row = waveM*16 + l16; K chunks [kk+8h .. +7], [kk+16+8h .. +7]
            const _Float16* ap = &Alds[(waveM * 16 + l16) * LDSW + kk + half * 8];
            v8h alo = *(const v8h*)ap;
            v8h ahi = *(const v8h*)(ap + 16);
            v16h afrag = __builtin_shufflevector(alo, ahi,
                0,1,2,3,4,5,6,7,8,9,10,11,12,13,14,15);

            #pragma unroll
            for (int f = 0; f < 4; ++f) {
                const int n = waveN * 64 + f * 16 + l16;
                const _Float16* bp = &Blds[n * LDSW + kk + half * 8];
                v8h blo = *(const v8h*)bp;
                v8h bhi = *(const v8h*)(bp + 16);
                v16h bfrag = __builtin_shufflevector(blo, bhi,
                    0,1,2,3,4,5,6,7,8,9,10,11,12,13,14,15);
                acc[f] = __builtin_amdgcn_wmma_f32_16x16x32_f16(
                    /*neg_a=*/false, afrag, /*neg_b=*/false, bfrag,
                    /*c_mod=*/(short)0, acc[f], /*reuse_a=*/false, /*reuse_b=*/false);
            }
        }

        // epilogue: bias + exact erf-GELU, store fp32
        #pragma unroll
        for (int f = 0; f < 4; ++f) {
            const int gcol = nc * N_CHUNK + waveN * 64 + f * 16 + l16;
            const float bv = bias[gcol];
            const int gr0  = rowBase + waveM * 16 + half * 8;
            #pragma unroll
            for (int r = 0; r < 8; ++r) {
                const float y = acc[f][r] + bv;
                const float g = 0.5f * y * (1.0f + erff(y * 0.7071067811865475f));
                out[(long)(gr0 + r) * F_DIM + gcol] = g;
            }
        }
    }
}

extern "C" void kernel_launch(void* const* d_in, const int* in_sizes, int n_in,
                              void* d_out, int out_size, void* d_ws, size_t ws_size,
                              hipStream_t stream) {
    const float* x399 = (const float*)d_in[0];
    const float* x365 = (const float*)d_in[1];
    const float* lng  = (const float*)d_in[2];
    const float* lnb  = (const float*)d_in[3];
    const float* W    = (const float*)d_in[4];
    const float* bias = (const float*)d_in[5];
    float* out = (float*)d_out;

    const int M = in_sizes[0] / D_DIM;   // B*S = 131072
    dim3 grid(M / M_TILE);               // 2048 blocks
    dim3 block(256);                     // 8 waves (wave32)
    fused_ln_gemm_gelu<<<grid, block, 0, stream>>>(x399, x365, lng, lnb, W, bias, out);
}
